// TransformerBlock_62474594287859
// MI455X (gfx1250) — matrix operand order repaired and checked by reading, not compiled, not run on hardware
//
#include <hip/hip_runtime.h>
#include <hip/hip_bf16.h>
#include <math.h>

// ---------------------------------------------------------------------------
// Types for CDNA5 WMMA
// ---------------------------------------------------------------------------
typedef __attribute__((ext_vector_type(16))) __bf16 v16bf;
typedef __attribute__((ext_vector_type(2)))  __bf16 v2bf;
typedef __attribute__((ext_vector_type(8)))  float  v8f;

// Hardware f32->bf16 converts (RTNE). Packed variant when clang exposes it.
static __device__ __forceinline__ __bf16 f2bf(float f) { return (__bf16)f; }

static __device__ __forceinline__ v2bf f2bf2(float x, float y) {
#if __has_builtin(__builtin_amdgcn_cvt_pk_bf16_f32)
    return __builtin_amdgcn_cvt_pk_bf16_f32(x, y);
#else
    v2bf r; r[0] = (__bf16)x; r[1] = (__bf16)y; return r;
#endif
}

#define D_MODEL 1024
#define SEQ     2048
#define N_HEADS 16
#define DH      64

// ---------------------------------------------------------------------------
// RMSNorm: one block per row of 1024
// ---------------------------------------------------------------------------
__global__ __launch_bounds__(256) void rmsnorm_kernel(
    const float* __restrict__ x, const float* __restrict__ scale,
    float* __restrict__ out)
{
    __shared__ float red[8];
    const int row = blockIdx.x;
    const float* xr = x + (size_t)row * D_MODEL;
    float ss = 0.0f;
    for (int i = threadIdx.x; i < D_MODEL; i += 256) {
        float f = xr[i];
        ss += f * f;
    }
    #pragma unroll
    for (int m = 1; m < 32; m <<= 1) ss += __shfl_xor(ss, m, 32);
    if ((threadIdx.x & 31) == 0) red[threadIdx.x >> 5] = ss;
    __syncthreads();
    float sum = 0.0f;
    #pragma unroll
    for (int i = 0; i < 8; ++i) sum += red[i];
    const float rn = rsqrtf(sum / (float)D_MODEL + 1e-8f);
    for (int i = threadIdx.x; i < D_MODEL; i += 256)
        out[(size_t)row * D_MODEL + i] = xr[i] * rn * scale[i];
}

// ---------------------------------------------------------------------------
// Generic GEMM: C[M,N] (+)= epilogue( A[M,K] * B[K,N] )  in bf16 WMMA, f32 acc
// Block tile 128x128, BK=32, 4 waves, wave tile 64x64 (4x4 WMMA tiles).
// Double-buffered LDS pipeline; 16 ds_load_b128 per 16 WMMAs per wave-step.
//   applySilu : silu(acc)
//   rowW      : acc *= rowW[row * rowWStride]   (nullptr to skip)
//   addSrc    : acc += addSrc[row*N+col]        (nullptr to skip)
//   accum     : acc += C_old
// ---------------------------------------------------------------------------
#define BM 128
#define BN 128
#define BK 32
#define LDP 40   // padded K-stride in LDS (bf16 elems)

__global__ __launch_bounds__(128) void gemm_bf16_wmma(
    const float* __restrict__ A, const float* __restrict__ B,
    float* __restrict__ C, int M, int N, int K,
    const float* __restrict__ addSrc,
    const float* __restrict__ rowW, int rowWStride,
    int accum, int applySilu)
{
    __shared__ __bf16 As[2][BM][LDP];   // [m][k]
    __shared__ __bf16 Bs[2][BN][LDP];   // transposed: [n][k]

    const int bm   = blockIdx.y * BM;
    const int bn   = blockIdx.x * BN;
    const int tid  = threadIdx.x;
    const int lane = tid & 31;
    const int wid  = tid >> 5;      // 0..3
    const int waveM = wid >> 1;     // 0..1  (64-row slice)
    const int waveN = wid & 1;      // 0..1  (64-col slice)
    const int lr = lane & 15;
    const int hi = lane >> 4;

    v8f acc[4][4];
    #pragma unroll
    for (int i = 0; i < 4; ++i)
        #pragma unroll
        for (int j = 0; j < 4; ++j)
            #pragma unroll
            for (int e = 0; e < 8; ++e) acc[i][j][e] = 0.0f;

    const int a_tk = (tid & 7) * 4;   // 0..28
    const int a_tm = tid >> 3;        // 0..15  (rows, +p*16)
    const int b_tn = (tid & 31) * 4;  // 0..124
    const int b_tk = tid >> 5;        // 0..3   (k, +p*4)

    // ---- stage tile kt=0 into buffer 0 ----
    #pragma unroll
    for (int p = 0; p < 8; ++p) {
        const int m = a_tm + p * 16;
        const float4 f = *(const float4*)(A + (size_t)(bm + m) * K + a_tk);
        *(v2bf*)&As[0][m][a_tk + 0] = f2bf2(f.x, f.y);
        *(v2bf*)&As[0][m][a_tk + 2] = f2bf2(f.z, f.w);
        const int kk = b_tk + p * 4;
        const float4 g = *(const float4*)(B + (size_t)kk * N + bn + b_tn);
        Bs[0][b_tn + 0][kk] = f2bf(g.x);
        Bs[0][b_tn + 1][kk] = f2bf(g.y);
        Bs[0][b_tn + 2][kk] = f2bf(g.z);
        Bs[0][b_tn + 3][kk] = f2bf(g.w);
    }
    __syncthreads();

    int buf = 0;
    for (int kt = 0; kt < K; kt += BK) {
        const bool has_next = (kt + BK) < K;
        float4 fa[8], fb[8];
        if (has_next) {
            // issue next tile's global loads early (latency hides under WMMA)
            #pragma unroll
            for (int p = 0; p < 8; ++p) {
                fa[p] = *(const float4*)(A + (size_t)(bm + a_tm + p * 16) * K + (kt + BK) + a_tk);
                fb[p] = *(const float4*)(B + (size_t)(kt + BK + b_tk + p * 4) * N + bn + b_tn);
            }
            if (kt + 2 * BK < K) {
                __builtin_prefetch(A + (size_t)(bm + a_tm) * K + (kt + 2 * BK) + a_tk, 0, 1);
                __builtin_prefetch(B + (size_t)(kt + 2 * BK + b_tk) * N + bn + b_tn, 0, 1);
            }
        }

        // ---- fragments per ISA 16-bit A (16x32) / B (32x16) layouts ----
        v16bf bF[4];
        #pragma unroll
        for (int ni = 0; ni < 4; ++ni) {
            const int col = waveN * 64 + ni * 16 + lr;
            #pragma unroll
            for (int j = 0; j < 8; ++j) {
                const int kk = ((j < 4) ? 2 * j : 16 + 2 * (j - 4)) + 8 * hi;
                bF[ni][2 * j]     = Bs[buf][col][kk];
                bF[ni][2 * j + 1] = Bs[buf][col][kk + 1];
            }
        }
        #pragma unroll
        for (int mi = 0; mi < 4; ++mi) {
            v16bf aF;
            const int row = waveM * 64 + mi * 16 + lr;
            #pragma unroll
            for (int j = 0; j < 8; ++j) {
                const int kk = ((j < 4) ? 2 * j : 16 + 2 * (j - 4)) + 8 * hi;
                aF[2 * j]     = As[buf][row][kk];
                aF[2 * j + 1] = As[buf][row][kk + 1];
            }
            #pragma unroll
            for (int ni = 0; ni < 4; ++ni)
                acc[mi][ni] = __builtin_amdgcn_wmma_f32_16x16x32_bf16(
                    false, aF, false, bF[ni], (short)0, acc[mi][ni], false, false);
        }

        if (has_next) {
            const int nb = buf ^ 1;
            #pragma unroll
            for (int p = 0; p < 8; ++p) {
                const int m = a_tm + p * 16;
                *(v2bf*)&As[nb][m][a_tk + 0] = f2bf2(fa[p].x, fa[p].y);
                *(v2bf*)&As[nb][m][a_tk + 2] = f2bf2(fa[p].z, fa[p].w);
                const int kk = b_tk + p * 4;
                Bs[nb][b_tn + 0][kk] = f2bf(fb[p].x);
                Bs[nb][b_tn + 1][kk] = f2bf(fb[p].y);
                Bs[nb][b_tn + 2][kk] = f2bf(fb[p].z);
                Bs[nb][b_tn + 3][kk] = f2bf(fb[p].w);
            }
        }
        __syncthreads();
        buf ^= 1;
    }

    // ---- epilogue (C/D layout: VGPR i -> row i + 8*hi, lane lr -> col) ----
    #pragma unroll
    for (int mi = 0; mi < 4; ++mi)
        #pragma unroll
        for (int ni = 0; ni < 4; ++ni)
            #pragma unroll
            for (int i = 0; i < 8; ++i) {
                const int row = bm + waveM * 64 + mi * 16 + i + 8 * hi;
                const int col = bn + waveN * 64 + ni * 16 + lr;
                float v = acc[mi][ni][i];
                if (applySilu) v = v * __builtin_amdgcn_rcpf(1.0f + __expf(-v));
                if (rowW) v *= rowW[(size_t)row * rowWStride];
                const size_t idx = (size_t)row * N + col;
                float r = v;
                if (addSrc) r += addSrc[idx];
                if (accum)  r += C[idx];
                C[idx] = r;
            }
}

// ---------------------------------------------------------------------------
// RoPE in-place on (S, H, DH) fp32; one thread per (s, head, i<32)
// ---------------------------------------------------------------------------
__global__ __launch_bounds__(256) void rope_kernel(float* __restrict__ x)
{
    const int idx = blockIdx.x * blockDim.x + threadIdx.x;
    if (idx >= SEQ * N_HEADS * 32) return;
    const int i    = idx & 31;
    const int head = (idx >> 5) & 15;
    const int s    = idx >> 9;
    const float freq = __expf(-(float)i * (9.21034037f / 32.0f)); // 10000^(-i/32)
    const float ang  = (float)s * freq;
    float sn, cs;
    __sincosf(ang, &sn, &cs);
    const size_t base = (size_t)s * D_MODEL + head * DH + i;
    const float x1 = x[base], x2 = x[base + 32];
    x[base]      = x1 * cs - x2 * sn;
    x[base + 32] = x1 * sn + x2 * cs;
}

// ---------------------------------------------------------------------------
// WMMA flash attention. Grid: (S/64, H). 128 threads = 4 waves; wave w owns
// query rows q0 = qb*64 + w*16 .. +15. Causal, scale 1/8 folded into Q.
// K/V chunks staged cooperatively in LDS (V transposed for B-fragments).
// ---------------------------------------------------------------------------
__global__ __launch_bounds__(128) void mla_attention(
    const float* __restrict__ q, const float* __restrict__ k,
    const float* __restrict__ v, float* __restrict__ out)
{
    __shared__ __bf16 Ks[32][72];        // [key][dh]   (32-key chunk)
    __shared__ __bf16 Vt[64][44];        // [dh][key]   (transposed)
    __shared__ __bf16 pbuf[4][16][40];   // per-wave P transpose buffer

    const int head = blockIdx.y;
    const int qb   = blockIdx.x;
    const int wid  = threadIdx.x >> 5;
    const int lane = threadIdx.x & 31;
    const int lr = lane & 15, hi = lane >> 4;
    const int q0 = qb * 64 + wid * 16;
    const int HB = head * DH;

    // Q as two 16x32 bf16 A-fragments (Dh 0..31, 32..63), pre-scaled
    v16bf qA[2];
    #pragma unroll
    for (int half = 0; half < 2; ++half) {
        const float* qr = q + (size_t)(q0 + lr) * D_MODEL + HB + half * 32;
        #pragma unroll
        for (int j = 0; j < 8; ++j) {
            const int kk = ((j < 4) ? 2 * j : 16 + 2 * (j - 4)) + 8 * hi;
            const v2bf pr = f2bf2(qr[kk] * 0.125f, qr[kk + 1] * 0.125f);
            qA[half][2 * j]     = pr[0];
            qA[half][2 * j + 1] = pr[1];
        }
    }

    v8f O[4];
    #pragma unroll
    for (int c = 0; c < 4; ++c)
        #pragma unroll
        for (int e = 0; e < 8; ++e) O[c][e] = 0.0f;
    float mst[8], lst[8];
    #pragma unroll
    for (int i = 0; i < 8; ++i) { mst[i] = -1e30f; lst[i] = 0.0f; }

    const int s_key  = threadIdx.x >> 2;        // 0..31
    const int s_dh0  = (threadIdx.x & 3) * 16;  // 0,16,32,48
    const int kend_block = qb * 64 + 64;        // block-uniform causal bound

    for (int kt = 0; kt < kend_block; kt += 32) {
        // ---- cooperative staging of K (row-major) and V (transposed) ----
        {
            const float* kr = k + (size_t)(kt + s_key) * D_MODEL + HB + s_dh0;
            const float* vr = v + (size_t)(kt + s_key) * D_MODEL + HB + s_dh0;
            #pragma unroll
            for (int p = 0; p < 4; ++p) {
                const float4 f = *(const float4*)(kr + p * 4);
                *(v2bf*)&Ks[s_key][s_dh0 + p * 4 + 0] = f2bf2(f.x, f.y);
                *(v2bf*)&Ks[s_key][s_dh0 + p * 4 + 2] = f2bf2(f.z, f.w);
                const float4 g = *(const float4*)(vr + p * 4);
                Vt[s_dh0 + p * 4 + 0][s_key] = f2bf(g.x);
                Vt[s_dh0 + p * 4 + 1][s_key] = f2bf(g.y);
                Vt[s_dh0 + p * 4 + 2][s_key] = f2bf(g.z);
                Vt[s_dh0 + p * 4 + 3][s_key] = f2bf(g.w);
            }
        }
        __syncthreads();

        if (kt < q0 + 16) {   // wave-uniform causal skip
            // ---- S = Q K^T for two 16-key tiles ----
            v8f sc[2];
            #pragma unroll
            for (int t = 0; t < 2; ++t) {
                v8f z;
                #pragma unroll
                for (int e = 0; e < 8; ++e) z[e] = 0.0f;
                #pragma unroll
                for (int half = 0; half < 2; ++half) {
                    v16bf kB;
                    #pragma unroll
                    for (int j = 0; j < 8; ++j) {
                        const int dh = ((j < 4) ? 2 * j : 16 + 2 * (j - 4)) + 8 * hi + half * 32;
                        kB[2 * j]     = Ks[t * 16 + lr][dh];
                        kB[2 * j + 1] = Ks[t * 16 + lr][dh + 1];
                    }
                    z = __builtin_amdgcn_wmma_f32_16x16x32_bf16(
                            false, qA[half], false, kB, (short)0, z, false, false);
                }
                sc[t] = z;
            }
            // ---- causal mask ----
            #pragma unroll
            for (int t = 0; t < 2; ++t)
                #pragma unroll
                for (int i = 0; i < 8; ++i) {
                    const int qrow = q0 + i + 8 * hi;
                    const int key  = kt + t * 16 + lr;
                    if (key > qrow) sc[t][i] = -1e30f;
                }
            // ---- online softmax (row reductions across 16-lane groups) ----
            float al[8];
            #pragma unroll
            for (int i = 0; i < 8; ++i) {
                float rm = fmaxf(sc[0][i], sc[1][i]);
                rm = fmaxf(rm, __shfl_xor(rm, 1, 32));
                rm = fmaxf(rm, __shfl_xor(rm, 2, 32));
                rm = fmaxf(rm, __shfl_xor(rm, 4, 32));
                rm = fmaxf(rm, __shfl_xor(rm, 8, 32));
                const float newm = fmaxf(mst[i], rm);
                al[i] = __expf(mst[i] - newm);
                mst[i] = newm;
                sc[0][i] = __expf(sc[0][i] - newm);
                sc[1][i] = __expf(sc[1][i] - newm);
                float s = sc[0][i] + sc[1][i];
                s += __shfl_xor(s, 1, 32);
                s += __shfl_xor(s, 2, 32);
                s += __shfl_xor(s, 4, 32);
                s += __shfl_xor(s, 8, 32);
                lst[i] = lst[i] * al[i] + s;
            }
            #pragma unroll
            for (int c = 0; c < 4; ++c)
                #pragma unroll
                for (int i = 0; i < 8; ++i) O[c][i] *= al[i];
            // ---- transpose P (C-layout -> A-fragment layout) through LDS ----
            #pragma unroll
            for (int t = 0; t < 2; ++t)
                #pragma unroll
                for (int i = 0; i < 8; ++i)
                    pbuf[wid][i + 8 * hi][t * 16 + lr] = f2bf(sc[t][i]);
            v16bf pA;
            #pragma unroll
            for (int j = 0; j < 8; ++j) {
                const int kk = ((j < 4) ? 2 * j : 16 + 2 * (j - 4)) + 8 * hi;
                pA[2 * j]     = pbuf[wid][lr][kk];
                pA[2 * j + 1] = pbuf[wid][lr][kk + 1];
            }
            // ---- O += P V  (V chunk as four 32x16 B-fragments from Vt) ----
            #pragma unroll
            for (int c = 0; c < 4; ++c) {
                v16bf vB;
                const int dhc = c * 16 + lr;
                #pragma unroll
                for (int j = 0; j < 8; ++j) {
                    const int key = ((j < 4) ? 2 * j : 16 + 2 * (j - 4)) + 8 * hi;
                    vB[2 * j]     = Vt[dhc][key];
                    vB[2 * j + 1] = Vt[dhc][key + 1];
                }
                O[c] = __builtin_amdgcn_wmma_f32_16x16x32_bf16(
                            false, pA, false, vB, (short)0, O[c], false, false);
            }
        }
        __syncthreads();
    }
    // ---- write normalized output ----
    #pragma unroll
    for (int i = 0; i < 8; ++i) {
        const float inv = __builtin_amdgcn_rcpf(lst[i]);
        const int row = q0 + i + 8 * hi;
        #pragma unroll
        for (int c = 0; c < 4; ++c)
            out[(size_t)row * D_MODEL + HB + c * 16 + lr] = O[c][i] * inv;
    }
}

// ---------------------------------------------------------------------------
// Gating: softmax over 8 logits, top-2, renormalize -> dense_w (S,8)
// ---------------------------------------------------------------------------
__global__ __launch_bounds__(256) void gating_kernel(
    const float* __restrict__ xn, const float* __restrict__ Wg,
    float* __restrict__ dw)
{
    const int t = blockIdx.x * blockDim.x + threadIdx.x;
    if (t >= SEQ) return;
    const float* xr = xn + (size_t)t * D_MODEL;
    float lg[8];
    #pragma unroll
    for (int e = 0; e < 8; ++e) lg[e] = 0.0f;
    for (int i = 0; i < D_MODEL; ++i) {
        const float xv = xr[i];
        #pragma unroll
        for (int e = 0; e < 8; ++e) lg[e] += xv * Wg[i * 8 + e];
    }
    float mx = lg[0];
    #pragma unroll
    for (int e = 1; e < 8; ++e) mx = fmaxf(mx, lg[e]);
    float p[8];
    #pragma unroll
    for (int e = 0; e < 8; ++e) p[e] = __expf(lg[e] - mx);
    int i1 = 0;
    #pragma unroll
    for (int e = 1; e < 8; ++e) if (p[e] > p[i1]) i1 = e;
    int i2 = (i1 == 0) ? 1 : 0;
    #pragma unroll
    for (int e = 0; e < 8; ++e) if (e != i1 && p[e] > p[i2]) i2 = e;
    const float inv = __builtin_amdgcn_rcpf(p[i1] + p[i2]);
    #pragma unroll
    for (int e = 0; e < 8; ++e)
        dw[(size_t)t * 8 + e] = (e == i1) ? p[i1] * inv
                              : (e == i2) ? p[i2] * inv : 0.0f;
}

// ---------------------------------------------------------------------------
// Host-side orchestration
// ---------------------------------------------------------------------------
static inline void launch_gemm(const float* A, const float* B, float* C,
                               int M, int N, int K,
                               const float* addSrc, const float* rowW, int rws,
                               int accum, int silu, hipStream_t s)
{
    dim3 grid(N / BN, M / BM);
    gemm_bf16_wmma<<<grid, 128, 0, s>>>(A, B, C, M, N, K, addSrc, rowW, rws, accum, silu);
}

extern "C" void kernel_launch(void* const* d_in, const int* in_sizes, int n_in,
                              void* d_out, int out_size, void* d_ws, size_t ws_size,
                              hipStream_t stream)
{
    const float* x          = (const float*)d_in[0];
    const float* attn_scale = (const float*)d_in[1];
    const float* Wq         = (const float*)d_in[2];
    const float* Wdkv       = (const float*)d_in[3];
    const float* Wuk        = (const float*)d_in[4];
    const float* Wuv        = (const float*)d_in[5];
    const float* Wo         = (const float*)d_in[6];
    const float* moe_scale  = (const float*)d_in[7];
    const float* Wg         = (const float*)d_in[8];
    const float* Ws1        = (const float*)d_in[9];
    const float* Ws2        = (const float*)d_in[10];
    const float* Wr1        = (const float*)d_in[11];
    const float* Wr2        = (const float*)d_in[12];
    float* out = (float*)d_out;

    float* W = (float*)d_ws;
    size_t o = 0;
    float* xn  = W + o; o += (size_t)SEQ * D_MODEL;
    float* qb  = W + o; o += (size_t)SEQ * D_MODEL;
    float* cb  = W + o; o += (size_t)SEQ * 256;
    float* kb  = W + o; o += (size_t)SEQ * D_MODEL;
    float* vb  = W + o; o += (size_t)SEQ * D_MODEL;
    float* ao  = W + o; o += (size_t)SEQ * D_MODEL;
    float* hb  = W + o; o += (size_t)SEQ * D_MODEL;
    float* hn  = W + o; o += (size_t)SEQ * D_MODEL;
    float* hid = W + o; o += (size_t)SEQ * D_MODEL;
    float* dw  = W + o; o += (size_t)SEQ * 8;

    // ---- MLA ----
    rmsnorm_kernel<<<SEQ, 256, 0, stream>>>(x, attn_scale, xn);
    launch_gemm(xn, Wq,   qb, SEQ, 1024, 1024, nullptr, nullptr, 0, 0, 0, stream);
    launch_gemm(xn, Wdkv, cb, SEQ,  256, 1024, nullptr, nullptr, 0, 0, 0, stream);
    launch_gemm(cb, Wuk,  kb, SEQ, 1024,  256, nullptr, nullptr, 0, 0, 0, stream);
    launch_gemm(cb, Wuv,  vb, SEQ, 1024,  256, nullptr, nullptr, 0, 0, 0, stream);
    {
        const int nth = SEQ * N_HEADS * 32;
        rope_kernel<<<(nth + 255) / 256, 256, 0, stream>>>(qb);
        rope_kernel<<<(nth + 255) / 256, 256, 0, stream>>>(kb);
    }
    mla_attention<<<dim3(SEQ / 64, N_HEADS), 128, 0, stream>>>(qb, kb, vb, ao);
    // h = x + attn @ Wo
    launch_gemm(ao, Wo, hb, SEQ, 1024, 1024, x, nullptr, 0, 0, 0, stream);

    // ---- MoE ----
    rmsnorm_kernel<<<SEQ, 256, 0, stream>>>(hb, moe_scale, hn);
    gating_kernel<<<SEQ / 256, 256, 0, stream>>>(hn, Wg, dw);

    // shared experts: out = h + sum_s silu(hn Ws1[s]) Ws2[s]
    for (int s = 0; s < 2; ++s) {
        launch_gemm(hn, Ws1 + (size_t)s * 1024 * 1024, hid, SEQ, 1024, 1024,
                    nullptr, nullptr, 0, 0, 1, stream);
        launch_gemm(hid, Ws2 + (size_t)s * 1024 * 1024, out, SEQ, 1024, 1024,
                    (s == 0) ? hb : nullptr, nullptr, 0, (s == 0) ? 0 : 1, 0, stream);
    }
    // routed experts: out += dense_w[:,e] * ( silu(hn Wr1[e]) Wr2[e] )
    for (int e = 0; e < 8; ++e) {
        launch_gemm(hn, Wr1 + (size_t)e * 1024 * 1024, hid, SEQ, 1024, 1024,
                    nullptr, nullptr, 0, 0, 1, stream);
        launch_gemm(hid, Wr2 + (size_t)e * 1024 * 1024, out, SEQ, 1024, 1024,
                    nullptr, dw + e, 8, 1, 0, stream);
    }
}
